// ScaledDotProductAttention_36910948942463
// MI455X (gfx1250) — matrix operand order repaired and checked by reading, not compiled, not run on hardware
//
#include <hip/hip_runtime.h>
#include <math.h>

// CDNA5 / gfx1250 flash-attention (causal), bf16 WMMA + f32 accumulation.
// Wave32: each wave owns a 16-query tile; 8 waves/WG share 64-key K/V tiles
// staged in LDS as bf16 (V transposed). Inner block body is one straight-line
// basic block (no per-tile gating): masked elements yield exact zeros in P,
// so unconditionally computing all 4 score tiles is correct and lets the
// scheduler overlap ds_load_b128 latency with the WMMA pipe.

typedef __attribute__((ext_vector_type(16))) __bf16 v16bf;
typedef __attribute__((ext_vector_type(8)))  float  v8f;

#define WAVES 8
#define BQ    (WAVES * 16)   // 128 queries per workgroup
#define BK    64             // keys per inner block
#define DH    64             // head dim

__device__ __forceinline__ unsigned short f2bf(float f) {
    unsigned u = __builtin_bit_cast(unsigned, f);
    u += 0x8000u;                               // round-half-up (cheap)
    return (unsigned short)(u >> 16);
}

__device__ __forceinline__ unsigned pk2bf(float a, float b) {
#if __has_builtin(__builtin_amdgcn_cvt_pk_bf16_f32)
    auto r = __builtin_amdgcn_cvt_pk_bf16_f32(a, b);   // v_cvt_pk_bf16_f32
    return __builtin_bit_cast(unsigned, r);
#else
    unsigned ua = __builtin_bit_cast(unsigned, a) + 0x8000u;
    unsigned ub = __builtin_bit_cast(unsigned, b) + 0x8000u;
    return __builtin_amdgcn_perm(ub, ua, 0x07060302u); // {ub.hi16, ua.hi16}
#endif
}

__device__ __forceinline__ float fexp2(float x) {
#if __has_builtin(__builtin_amdgcn_exp2f)
    return __builtin_amdgcn_exp2f(x);                  // bare v_exp_f32
#else
    return __builtin_exp2f(x);
#endif
}

union AF {                   // WMMA bf16 operand fragment (16 halves / 8 VGPRs)
    v16bf v;
    unsigned u32[8];
    unsigned short u16[16];
};

__global__ __launch_bounds__(256, 1)
void fa_fwd_gfx1250(const float* __restrict__ Q, const float* __restrict__ K,
                    const float* __restrict__ V, float* __restrict__ O, int S) {
    __shared__ unsigned short Ks[BK * DH];        // bf16 [key][d]
    __shared__ unsigned short Vt[DH * BK];        // bf16 [d][key] (transposed)
    __shared__ unsigned short Ps[WAVES][16 * BK]; // bf16 P tiles, per wave

    const int tid  = threadIdx.x;
    const int lane = tid & 31;
    const int half = (lane >> 4) & 1;             // 0: lanes 0-15, 1: lanes 16-31
    const int ln   = lane & 15;

    // wave-uniform values -> SGPRs: scalar control flow, scalar LDS bases
    const int wave  = __builtin_amdgcn_readfirstlane(tid >> 5);
    const int bh    = blockIdx.y;
    const int qWG   = blockIdx.x * BQ;
    const int qBase = __builtin_amdgcn_readfirstlane(qWG + wave * 16);

    const float* Qh = Q + (size_t)bh * S * DH;
    const float* Kh = K + (size_t)bh * S * DH;
    const float* Vh = V + (size_t)bh * S * DH;
    float*       Oh = O + (size_t)bh * S * DH;

    // 1/sqrt(64) * log2(e): softmax computed in base-2 domain
    const float scale = 0.125f * 1.44269504088896f;

    // ---- Q A-fragments (two 32-wide K-pieces along D), built once ----
    AF aQ0, aQ1;
    {
        const float* qrow = Qh + (size_t)(qBase + ln) * DH;
        #pragma unroll
        for (int g = 0; g < 8; ++g) {
            int dk = half * 16 + g * 2;
            aQ0.u32[g] = pk2bf(qrow[dk] * scale,      qrow[dk + 1] * scale);
            aQ1.u32[g] = pk2bf(qrow[32 + dk] * scale, qrow[33 + dk] * scale);
        }
    }

    // ---- accumulators: O as 4 C-fragments over d-tiles; softmax state ----
    v8f o0 = {}, o1 = {}, o2 = {}, o3 = {};
    float m_i[8], l_i[8];
    #pragma unroll
    for (int g = 0; g < 8; ++g) { m_i[g] = -1e30f; l_i[g] = 0.0f; }

    const int kEndWG   = qWG + BQ;                // WG-wide causal key bound
    const int kEndWave = qBase + 16;              // this wave's bound (scalar)
    const int cBase    = ln - half * 8 - qBase;   // per-lane mask offset

    // cooperative-fill coordinates: 256 threads; each owns 2 keys x 8 dims
    const int key = (tid >> 3) * 2;
    const int d0  = (tid & 7) * 8;

    for (int kb = 0; kb < kEndWG; kb += BK) {
        __syncthreads();                           // prior LDS reads done
        {
            const float* kr0 = Kh + (size_t)(kb + key) * DH + d0;
            const float* kr1 = kr0 + DH;
            const float* vr0 = Vh + (size_t)(kb + key) * DH + d0;
            const float* vr1 = vr0 + DH;
            float4 a0 = ((const float4*)kr0)[0], a1 = ((const float4*)kr0)[1];
            float4 b0 = ((const float4*)kr1)[0], b1 = ((const float4*)kr1)[1];
            float4 c0 = ((const float4*)vr0)[0], c1 = ((const float4*)vr0)[1];
            float4 e0 = ((const float4*)vr1)[0], e1 = ((const float4*)vr1)[1];

            unsigned* kd0 = (unsigned*)&Ks[(key    ) * DH + d0];
            unsigned* kd1 = (unsigned*)&Ks[(key + 1) * DH + d0];
            kd0[0] = pk2bf(a0.x, a0.y); kd0[1] = pk2bf(a0.z, a0.w);
            kd0[2] = pk2bf(a1.x, a1.y); kd0[3] = pk2bf(a1.z, a1.w);
            kd1[0] = pk2bf(b0.x, b0.y); kd1[1] = pk2bf(b0.z, b0.w);
            kd1[2] = pk2bf(b1.x, b1.y); kd1[3] = pk2bf(b1.z, b1.w);

            float va[8] = {c0.x, c0.y, c0.z, c0.w, c1.x, c1.y, c1.z, c1.w};
            float vb[8] = {e0.x, e0.y, e0.z, e0.w, e1.x, e1.y, e1.z, e1.w};
            #pragma unroll
            for (int i = 0; i < 8; ++i)            // pack key-pair at fixed d
                *(unsigned*)&Vt[(d0 + i) * BK + key] = pk2bf(va[i], vb[i]);

            if (kb + BK < kEndWG) {                // gfx1250 global_prefetch_b8
                __builtin_prefetch(kr0 + (size_t)BK * DH, 0, 0);
                __builtin_prefetch(kr1 + (size_t)BK * DH, 0, 0);
                __builtin_prefetch(vr0 + (size_t)BK * DH, 0, 0);
                __builtin_prefetch(vr1 + (size_t)BK * DH, 0, 0);
            }
        }
        __syncthreads();                           // fill visible

        if (kb >= kEndWave) continue;              // scalar branch

        // ---- scores S = Q K^T : all 4 tiles, straight-line ----
        v8f s[4];
        #pragma unroll
        for (int t = 0; t < 4; ++t) {
            AF bk0, bk1;
            const unsigned* krow = (const unsigned*)&Ks[(t * 16 + ln) * DH];
            #pragma unroll
            for (int g = 0; g < 8; ++g) {
                bk0.u32[g] = krow[half * 8 + g];        // d = (half?16:0)+2g
                bk1.u32[g] = krow[16 + half * 8 + g];   // d = 32 + ...
            }
            v8f acc = {};
            acc = __builtin_amdgcn_wmma_f32_16x16x32_bf16(false, aQ0.v, false, bk0.v,
                                                          (short)0, acc, false, false);
            acc = __builtin_amdgcn_wmma_f32_16x16x32_bf16(false, aQ1.v, false, bk1.v,
                                                          (short)0, acc, false, false);
            s[t] = acc;
        }

        // ---- causal mask (diagonal block only; scalar gate) ----
        // element (t,g) masked iff g < kb + 16*t + cBase; masked -> p == 0,
        // so tiles fully past the causal limit contribute nothing.
        if (kb + BK - 1 > qBase) {
            #pragma unroll
            for (int t = 0; t < 4; ++t) {
                int lim = kb + 16 * t + cBase;
                #pragma unroll
                for (int g = 0; g < 8; ++g)
                    if (g < lim) s[t][g] = -1e30f;
            }
        }

        // ---- online softmax (rows split across 16-lane halves) ----
        float mx[8];
        #pragma unroll
        for (int g = 0; g < 8; ++g)
            mx[g] = fmaxf(fmaxf(s[0][g], s[1][g]), fmaxf(s[2][g], s[3][g]));
        #pragma unroll
        for (int off = 1; off < 16; off <<= 1) {
            #pragma unroll
            for (int g = 0; g < 8; ++g)
                mx[g] = fmaxf(mx[g], __shfl_xor(mx[g], off, 32));
        }

        float al[8];
        #pragma unroll
        for (int g = 0; g < 8; ++g) {
            float mn = fmaxf(m_i[g], mx[g]);
            al[g]  = fexp2(m_i[g] - mn);
            m_i[g] = mn;
        }

        // ---- P = exp2(S - m): convert + stash in LDS, accumulate row sums ----
        unsigned short* pw = Ps[wave];
        float rs[8];
        #pragma unroll
        for (int g = 0; g < 8; ++g) rs[g] = 0.0f;
        #pragma unroll
        for (int t = 0; t < 4; ++t) {
            #pragma unroll
            for (int g = 0; g < 8; ++g) {
                float pv = fexp2(s[t][g] - m_i[g]);
                rs[g] += pv;
                pw[(g + half * 8) * BK + 16 * t + ln] = f2bf(pv);
            }
        }
        #pragma unroll
        for (int off = 1; off < 16; off <<= 1) {
            #pragma unroll
            for (int g = 0; g < 8; ++g)
                rs[g] += __shfl_xor(rs[g], off, 32);
        }
        #pragma unroll
        for (int g = 0; g < 8; ++g) {
            l_i[g] = l_i[g] * al[g] + rs[g];
            o0[g] *= al[g]; o1[g] *= al[g]; o2[g] *= al[g]; o3[g] *= al[g];
        }

        // ---- O += P V : two 32-key pieces x four 16-wide d-tiles ----
        const unsigned* prow = (const unsigned*)&pw[ln * BK];
        #pragma unroll
        for (int p = 0; p < 2; ++p) {
            AF aP;
            #pragma unroll
            for (int g = 0; g < 8; ++g)
                aP.u32[g] = prow[16 * p + half * 8 + g];
            #pragma unroll
            for (int t = 0; t < 4; ++t) {
                AF bV;
                const unsigned* vrow = (const unsigned*)&Vt[(t * 16 + ln) * BK];
                #pragma unroll
                for (int g = 0; g < 8; ++g)
                    bV.u32[g] = vrow[16 * p + half * 8 + g]; // key=32p+(half?16:0)+2g
                v8f acc = (t == 0) ? o0 : (t == 1) ? o1 : (t == 2) ? o2 : o3;
                acc = __builtin_amdgcn_wmma_f32_16x16x32_bf16(false, aP.v, false, bV.v,
                                                              (short)0, acc, false, false);
                if (t == 0) o0 = acc; else if (t == 1) o1 = acc;
                else if (t == 2) o2 = acc; else o3 = acc;
            }
        }
    }

    // ---- finalize: O /= l, write fp32 ----
    #pragma unroll
    for (int g = 0; g < 8; ++g) {
        float inv = 1.0f / l_i[g];
        int row = g + half * 8;
        float* orow = Oh + (size_t)(qBase + row) * DH + ln;
        orow[0]  = o0[g] * inv;
        orow[16] = o1[g] * inv;
        orow[32] = o2[g] * inv;
        orow[48] = o3[g] * inv;
    }
}

extern "C" void kernel_launch(void* const* d_in, const int* in_sizes, int n_in,
                              void* d_out, int out_size, void* d_ws, size_t ws_size,
                              hipStream_t stream) {
    (void)n_in; (void)d_ws; (void)ws_size; (void)out_size;
    const float* Q = (const float*)d_in[0];
    const float* K = (const float*)d_in[1];
    const float* V = (const float*)d_in[2];
    // d_in[3] is the causal mask; causality is applied analytically.
    float* O = (float*)d_out;

    int S  = (int)(sqrt((double)in_sizes[3]) + 0.5);   // mask is S*S
    int BH = in_sizes[0] / (S * DH);                   // B*H

    dim3 grid(S / BQ, BH);
    fa_fwd_gfx1250<<<grid, 256, 0, stream>>>(Q, K, V, O, S);
}
